// ReActNet_BasicBlock_5239860101351
// MI455X (gfx1250) — 8-bit matrix operand order repaired and checked by reading, compiled, not run on hardware
//
#include <hip/hip_runtime.h>
#include <cstdint>

typedef int v8i __attribute__((ext_vector_type(8)));

#define NPIX   50176
#define NCH    256
#define KDIM   2304
#define HW     784
#define NIMG   64
#define PW     30
#define PIMG   900
#define NPT    392

__global__ __launch_bounds__(256) void zerofill_kernel(int4* __restrict__ p) {
  const int4 z = {0, 0, 0, 0};
  p[(size_t)blockIdx.x * 256 + threadIdx.x] = z;
}

__global__ __launch_bounds__(256) void prepw_kernel(
    const float* __restrict__ w, int8_t* __restrict__ signW,
    float* __restrict__ scale)
{
  const int p = blockIdx.x;
  const int c = threadIdx.x;
  const float* wp = w + ((size_t)p * NCH + c) * 9;
  float asum = 0.f;
#pragma unroll
  for (int t = 0; t < 9; ++t) {
    float v = wp[t];
    asum += fabsf(v);
    int8_t s = (v > 0.f) ? 1 : ((v < 0.f) ? -1 : 0);
    signW[(size_t)p * KDIM + t * NCH + c] = s;
  }
  __shared__ float sh[256];
  sh[c] = asum;
  __syncthreads();
  for (int off = 128; off > 0; off >>= 1) {
    if (c < off) sh[c] += sh[c + off];
    __syncthreads();
  }
  if (c == 0) scale[p] = sh[0] * (1.0f / (float)KDIM);
}

__global__ __launch_bounds__(256) void signx_kernel(
    const float* __restrict__ x, const float* __restrict__ b10,
    int8_t* __restrict__ signXp)
{
  __shared__ int8_t sm[64][64];
  const int n = blockIdx.z, c0 = blockIdx.y * 64, hw0 = blockIdx.x * 64;
  const int tid = threadIdx.x;
  const int lhw = tid & 63, lc4 = tid >> 6;
#pragma unroll 4
  for (int i = 0; i < 16; ++i) {
    const int lc = lc4 + i * 4;
    const int c = c0 + lc;
    const int hw = hw0 + lhw;
    int8_t s = 0;
    if (hw < HW) {
      float v = x[((size_t)n * NCH + c) * HW + hw] + b10[c];
      s = (v > 0.f) ? 1 : ((v < 0.f) ? -1 : 0);
    }
    sm[lhw][lc] = s;
  }
  __syncthreads();
  const int wc = tid & 63, whw4 = tid >> 6;
#pragma unroll 4
  for (int i = 0; i < 16; ++i) {
    const int whw = whw4 + i * 4;
    const int hw = hw0 + whw;
    if (hw < HW) {
      const int h = hw / 28, w = hw % 28;
      signXp[((size_t)n * PIMG + (h + 1) * PW + (w + 1)) * NCH + c0 + wc] =
          sm[whw][wc];
    }
  }
}

struct Slab { int4 a0, a1, b0, b1; };

__device__ __forceinline__ v8i ldA(const int8_t* p) {
  v8i a;
  a[0] = *(const int*)(p + 0);  a[1] = *(const int*)(p + 4);
  a[2] = *(const int*)(p + 8);  a[3] = *(const int*)(p + 12);
  a[4] = *(const int*)(p + 32); a[5] = *(const int*)(p + 36);
  a[6] = *(const int*)(p + 40); a[7] = *(const int*)(p + 44);
  return a;
}
__device__ __forceinline__ v8i ldB(const int8_t* p) {
  int4 lo = *(const int4*)(p);
  int4 hi = *(const int4*)(p + 32);
  v8i b = {lo.x, lo.y, lo.z, lo.w, hi.x, hi.y, hi.z, hi.w};
  return b;
}

__global__ __launch_bounds__(256) void bconv_kernel(
    const int8_t* __restrict__ signXp,
    const int8_t* __restrict__ signW,
    const float* __restrict__ scale,
    float* __restrict__ out,
    int2* __restrict__ part)
{
  __shared__ alignas(16) int8_t Asm[2 * 128 * 64];
  __shared__ alignas(16) int8_t Bsm[2 * 128 * 64];
  __shared__ int cs1[128];
  __shared__ int cs2[128];

  const int pixBase = blockIdx.x * 128;
  const int pBase   = blockIdx.y * 128;
  const int tid = threadIdx.x;
  if (tid < 128) { cs1[tid] = 0; cs2[tid] = 0; }

  const int rj = tid >> 1;
  const int rq = (tid & 1) * 32;
  const int8_t* aGlob = signW + (size_t)(pBase + rj) * KDIM + rq;

  const int pix  = pixBase + rj;
  const int nImg = pix / HW;
  const int hw   = pix % HW;
  const int hh   = hw / 28, ww = hw % 28;
  const int8_t* bCenter =
      signXp + ((size_t)nImg * PIMG + (hh + 1) * PW + (ww + 1)) * NCH + rq;

  const int lane = tid & 31;
  const int wid  = tid >> 5;
  const int mG   = (wid & 1) * 64;
  const int nG   = (wid >> 1) * 32;

  const int am  = lane & 15;
  const int akb = (lane & 16) ? 16 : 0;
  const int bn  = lane & 15;
  const int bkb = (lane & 16) ? 16 : 0;

  const v8i vzero = {0, 0, 0, 0, 0, 0, 0, 0};
  v8i acc[4][2];
#pragma unroll
  for (int i = 0; i < 4; ++i)
#pragma unroll
    for (int j = 0; j < 2; ++j) acc[i][j] = vzero;

  auto fetch = [&](int s) -> Slab {
    Slab r;
    r.a0 = *(const int4*)(aGlob + s * 64);
    r.a1 = *(const int4*)(aGlob + s * 64 + 16);
    const int tap = s >> 2, cq = s & 3;
    const int soff = ((tap / 3 - 1) * PW + (tap % 3 - 1)) * NCH + cq * 64;
    r.b0 = *(const int4*)(bCenter + soff);
    r.b1 = *(const int4*)(bCenter + soff + 16);
    return r;
  };
  auto stage = [&](int buf, const Slab& sl) {
    int8_t* A = Asm + buf * (128 * 64);
    int8_t* B = Bsm + buf * (128 * 64);
    *(int4*)&A[rj * 64 + rq]      = sl.a0;
    *(int4*)&A[rj * 64 + rq + 16] = sl.a1;
    *(int4*)&B[rj * 64 + rq]      = sl.b0;
    *(int4*)&B[rj * 64 + rq + 16] = sl.b1;
  };

  { Slab s0 = fetch(0); stage(0, s0); }
  __syncthreads();

#pragma unroll
  for (int s = 0; s < 36; ++s) {
    const int buf = s & 1;
    const bool more = (s + 1) < 36;
    Slab nxt;
    if (more) nxt = fetch(s + 1);

    const int8_t* A = Asm + buf * (128 * 64);
    const int8_t* B = Bsm + buf * (128 * 64);
    v8i af[4], bf[2];
#pragma unroll
    for (int i = 0; i < 4; ++i)
      af[i] = ldA(A + (mG + i * 16 + am) * 64 + akb);
#pragma unroll
    for (int j = 0; j < 2; ++j)
      bf[j] = ldB(B + (nG + j * 16 + bn) * 64 + bkb);

#pragma unroll
    for (int i = 0; i < 4; ++i)
#pragma unroll
      for (int j = 0; j < 2; ++j)
        acc[i][j] = __builtin_amdgcn_wmma_i32_16x16x64_iu8(
            true, af[i], true, bf[j], acc[i][j], false, false);

    if (more) stage(buf ^ 1, nxt);
    __syncthreads();
  }

  const int col   = lane & 15;
  const int rowHi = (lane & 16) ? 8 : 0;
#pragma unroll
  for (int i = 0; i < 4; ++i) {
#pragma unroll
    for (int r = 0; r < 8; ++r) {
      const int v0 = acc[i][0][r];
      const int v1 = acc[i][1][r];
      const int m = pBase + mG + i * 16 + rowHi + r;
      const float sc = scale[m];
      const size_t nbase = (size_t)m * NPIX + pixBase + nG + col;
      out[nbase]      = sc * (float)v0;
      out[nbase + 16] = sc * (float)v1;

      int s1 = v0 + v1;
      int s2 = v0 * v0 + v1 * v1;
#pragma unroll
      for (int d = 1; d < 16; d <<= 1) {
        s1 += __shfl_xor(s1, d, 16);
        s2 += __shfl_xor(s2, d, 16);
      }
      if (col == 0) {
        const int cloc = mG + i * 16 + rowHi + r;
        atomicAdd(&cs1[cloc], s1);
        atomicAdd(&cs2[cloc], s2);
      }
    }
  }
  __syncthreads();
  if (tid < 128) {
    const int ch = pBase + tid;
    int2 v; v.x = cs1[tid]; v.y = cs2[tid];
    part[(size_t)ch * NPT + blockIdx.x] = v;
  }
}

__global__ __launch_bounds__(256) void finalize_kernel(
    const int2* __restrict__ part, const float* __restrict__ scale,
    float* __restrict__ mu, float* __restrict__ rinv)
{
  const int p = blockIdx.x;
  const int tid = threadIdx.x;
  long long s1 = 0, s2 = 0;
  for (int i = tid; i < NPT; i += 256) {
    int2 v = part[(size_t)p * NPT + i];
    s1 += v.x;
    s2 += v.y;
  }
  __shared__ long long sh1[256], sh2[256];
  sh1[tid] = s1; sh2[tid] = s2;
  __syncthreads();
  for (int off = 128; off > 0; off >>= 1) {
    if (tid < off) { sh1[tid] += sh1[tid + off]; sh2[tid] += sh2[tid + off]; }
    __syncthreads();
  }
  if (tid == 0) {
    const float sc = scale[p];
    const float m   = sc * (float)sh1[0] * (1.0f / (float)NPIX);
    const float ex2 = sc * sc * (float)sh2[0] * (1.0f / (float)NPIX);
    mu[p] = m;
    rinv[p] = rsqrtf(ex2 - m * m + 1e-5f);
  }
}

__global__ __launch_bounds__(256) void ew1_kernel(
    const float* __restrict__ conv, const float* __restrict__ x,
    const float* __restrict__ gamma1, const float* __restrict__ beta1,
    const float* __restrict__ b11, const float* __restrict__ a1,
    const float* __restrict__ b12, const float* __restrict__ b20,
    const float* __restrict__ mu, const float* __restrict__ rinv,
    float* __restrict__ hbuf, int8_t* __restrict__ signXp)
{
  __shared__ int8_t sm[64][64];
  const int n = blockIdx.z, c0 = blockIdx.y * 64, hw0 = blockIdx.x * 64;
  const int tid = threadIdx.x;
  const int lhw = tid & 63, lc4 = tid >> 6;
#pragma unroll 4
  for (int i = 0; i < 16; ++i) {
    const int lc = lc4 + i * 4;
    const int c = c0 + lc;
    const int hw = hw0 + lhw;
    int8_t s = 0;
    if (hw < HW) {
      const size_t pixI = (size_t)n * HW + hw;
      const size_t nchw = ((size_t)n * NCH + c) * HW + hw;
      float cv = conv[(size_t)c * NPIX + pixI];
      float bnv = gamma1[c] * (cv - mu[c]) * rinv[c] + beta1[c];
      float t = bnv + x[nchw] + b11[c];
      float hv = (t >= 0.f ? t : a1[c] * t) + b12[c];
      hbuf[nchw] = hv;
      float u = hv + b20[c];
      s = (u > 0.f) ? 1 : ((u < 0.f) ? -1 : 0);
    }
    sm[lhw][lc] = s;
  }
  __syncthreads();
  const int wc = tid & 63, whw4 = tid >> 6;
#pragma unroll 4
  for (int i = 0; i < 16; ++i) {
    const int whw = whw4 + i * 4;
    const int hw = hw0 + whw;
    if (hw < HW) {
      const int h = hw / 28, w = hw % 28;
      signXp[((size_t)n * PIMG + (h + 1) * PW + (w + 1)) * NCH + c0 + wc] =
          sm[whw][wc];
    }
  }
}

__global__ __launch_bounds__(256) void ew2_kernel(
    const float* __restrict__ conv, const float* __restrict__ hbuf,
    const float* __restrict__ gamma2, const float* __restrict__ beta2,
    const float* __restrict__ b21, const float* __restrict__ a2,
    const float* __restrict__ b22, const float* __restrict__ mu,
    const float* __restrict__ rinv, float* __restrict__ out)
{
  const size_t idx = (size_t)blockIdx.x * 256 + threadIdx.x;
  const int n = (int)(idx / (NCH * HW));
  const int rem = (int)(idx % (NCH * HW));
  const int p = rem / HW;
  const int hw = rem % HW;
  float cv = conv[(size_t)p * NPIX + (size_t)n * HW + hw];
  float bnv = gamma2[p] * (cv - mu[p]) * rinv[p] + beta2[p];
  float t = bnv + hbuf[idx] + b21[p];
  out[idx] = (t >= 0.f ? t : a2[p] * t) + b22[p];
}

extern "C" void kernel_launch(void* const* d_in, const int* in_sizes, int n_in,
                              void* d_out, int out_size, void* d_ws, size_t ws_size,
                              hipStream_t stream) {
  const float* x      = (const float*)d_in[0];
  const float* b10    = (const float*)d_in[1];
  const float* w1     = (const float*)d_in[2];
  const float* gamma1 = (const float*)d_in[3];
  const float* beta1  = (const float*)d_in[4];
  const float* b11    = (const float*)d_in[5];
  const float* a1     = (const float*)d_in[6];
  const float* b12    = (const float*)d_in[7];
  const float* b20    = (const float*)d_in[8];
  const float* w2     = (const float*)d_in[9];
  const float* gamma2 = (const float*)d_in[10];
  const float* beta2  = (const float*)d_in[11];
  const float* b21    = (const float*)d_in[12];
  const float* a2     = (const float*)d_in[13];
  const float* b22    = (const float*)d_in[14];
  float* out = (float*)d_out;

  char* ws = (char*)d_ws;
  const size_t TENS = (size_t)NPIX * NCH * sizeof(float);
  const size_t PADB = (size_t)NIMG * PIMG * NCH;
  float*  hbuf   = (float*)(ws);
  float*  convb  = (float*)(ws + TENS);
  int8_t* signXp = (int8_t*)(ws + 2 * TENS);
  int8_t* signW1 = signXp + PADB;
  int8_t* signW2 = signW1 + (size_t)NCH * KDIM;
  float*  scale1 = (float*)(signW2 + (size_t)NCH * KDIM);
  float*  scale2 = scale1 + NCH;
  float*  mu     = scale2 + NCH;
  float*  rinv   = mu + NCH;
  int2*   part   = (int2*)(rinv + NCH);

  const dim3 ewGrid(13, 4, NIMG);
  const dim3 convGrid(NPT, NCH / 128);
  const int  zeroBlocks = (int)(PADB / 16 / 256);

  prepw_kernel<<<NCH, 256, 0, stream>>>(w1, signW1, scale1);
  prepw_kernel<<<NCH, 256, 0, stream>>>(w2, signW2, scale2);
  zerofill_kernel<<<zeroBlocks, 256, 0, stream>>>((int4*)signXp);

  signx_kernel<<<ewGrid, 256, 0, stream>>>(x, b10, signXp);
  bconv_kernel<<<convGrid, 256, 0, stream>>>(signXp, signW1, scale1, convb, part);
  finalize_kernel<<<NCH, 256, 0, stream>>>(part, scale1, mu, rinv);
  ew1_kernel<<<ewGrid, 256, 0, stream>>>(convb, x, gamma1, beta1, b11, a1,
                                         b12, b20, mu, rinv, hbuf, signXp);

  bconv_kernel<<<convGrid, 256, 0, stream>>>(signXp, signW2, scale2, convb, part);
  finalize_kernel<<<NCH, 256, 0, stream>>>(part, scale2, mu, rinv);
  ew2_kernel<<<NPIX, 256, 0, stream>>>(convb, hbuf, gamma2, beta2, b21, a2,
                                       b22, mu, rinv, out);
}
